// MessagePassingConvolution_17935783428299
// MI455X (gfx1250) — matrix operand order repaired and checked by reading, not compiled, not run on hardware
//
#include <hip/hip_runtime.h>
#include <math.h>

// CDNA5 / gfx1250 WMMA vector types
typedef __attribute__((ext_vector_type(16))) _Float16 v16h;
typedef __attribute__((ext_vector_type(8)))  float    v8f;

#define CCH 128   // channels C

// silu(x) = x * sigmoid(x) = x * rcp(1 + exp(-x)).
// Use the hardware v_rcp_f32 (1-ulp class) instead of the IEEE divide
// expansion (v_div_scale + Newton) the precise '/' would emit.
__device__ __forceinline__ float silu_f(float x) {
  return x * __builtin_amdgcn_rcpf(1.0f + __expf(-x));
}

__global__ void zero_out_kernel(float* __restrict__ p, int n) {
  int i = blockIdx.x * blockDim.x + threadIdx.x;
  if (i < n) p[i] = 0.0f;
}

// One 256-thread block (8 wave32) handles a tile of 16 edges.
// Phase 1: radial MLP (R->64->64->64->640) on the 16 edges via
//          v_wmma_f32_16x16x32_f16; "mix" (16x640, with all normalization
//          scales folded in) is kept in LDS.
// Phase 2: 16 lanes per edge: gather sender feats (L2-resident), tensor
//          products, scale by mix, atomic-scatter into receiver rows.
__launch_bounds__(256)
__global__ void mpconv_kernel(
    const float* __restrict__ vectors,
    const float* __restrict__ node_feats,
    const float* __restrict__ radial,
    const float* __restrict__ w0,
    const float* __restrict__ w1,
    const float* __restrict__ w2,
    const float* __restrict__ w3,
    const int*   __restrict__ senders,
    const int*   __restrict__ receivers,
    float*       __restrict__ out,
    int E, int R)
{
  __shared__ _Float16 sH[8][16][64];    // per-wave hidden activations (f16) : 16 KB
  __shared__ float    sMix[16][640];    // mix for the 16-edge tile          : 40 KB

  const int tid   = threadIdx.x;
  const int wave  = tid >> 5;       // 0..7
  const int lane  = tid & 31;
  const int hlf   = lane >> 4;      // 0/1 : lane half (WMMA layout)
  const int m     = lane & 15;      // row (A/C) or col (B/C/D) index
  const int ebase = blockIdx.x * 16;

  // ------------------------------------------------------------------
  // Layer 0: radial (16 x R, zero-padded to K=32)  @  w0 (R x 64)
  // A layout: A[lane][e] : M = m,          K = e + 8*hlf + (e>=8 ? 8 : 0)
  // B layout: B[lane][e] : K = e + 8*hlf + (e>=8 ? 8 : 0), N = m
  // D layout: D[lane][r] : M = r + 8*hlf,  N = m
  // Guards are branchless (clamped address + select) so no EXEC churn.
  // ------------------------------------------------------------------
  {
    v16h a;
    const int  row    = ebase + m;
    const bool rowok  = row < E;
    const int  rbase  = rowok ? (row * R) : 0;
#pragma unroll
    for (int e = 0; e < 16; ++e) {
      const int  k  = e + 8 * hlf + ((e >= 8) ? 8 : 0);
      const bool ok = (k < R) & rowok;
      const float v = radial[ok ? (rbase + k) : 0];   // unconditional load
      a[e] = (_Float16)(ok ? v : 0.0f);
    }
    const float sc = rsqrtf((float)R);
#pragma unroll
    for (int nt = 0; nt < 4; ++nt) {
      v16h b;
#pragma unroll
      for (int e = 0; e < 16; ++e) {
        const int  k  = e + 8 * hlf + ((e >= 8) ? 8 : 0);
        const bool ok = (k < R);
        const float v = w0[ok ? (k * 64 + nt * 16 + m) : 0];
        b[e] = (_Float16)(ok ? v : 0.0f);
      }
      v8f c = {};
      c = __builtin_amdgcn_wmma_f32_16x16x32_f16(false, a, false, b, (short)0, c, false, false);
#pragma unroll
      for (int r = 0; r < 8; ++r) {
        float x = silu_f(c[r] * sc);
        sH[wave][r + 8 * hlf][nt * 16 + m] = (_Float16)x;
      }
    }
  }

  // ------------------------------------------------------------------
  // Layers 1,2: H (16x64) @ W (64x64), K split into two 32-chunks
  // (per-wave private LDS region; wave-internal DS ordering is in-order)
  // ------------------------------------------------------------------
#pragma unroll
  for (int layer = 0; layer < 2; ++layer) {
    const float* __restrict__ W = (layer == 0) ? w1 : w2;
    v16h a0, a1;
#pragma unroll
    for (int e = 0; e < 16; ++e) {
      const int k = e + 8 * hlf + ((e >= 8) ? 8 : 0);
      a0[e] = sH[wave][m][k];
      a1[e] = sH[wave][m][32 + k];
    }
#pragma unroll
    for (int nt = 0; nt < 4; ++nt) {
      v16h b0, b1;
#pragma unroll
      for (int e = 0; e < 16; ++e) {
        const int k = e + 8 * hlf + ((e >= 8) ? 8 : 0);
        b0[e] = (_Float16)W[k * 64 + nt * 16 + m];
        b1[e] = (_Float16)W[(k + 32) * 64 + nt * 16 + m];
      }
      v8f c = {};
      c = __builtin_amdgcn_wmma_f32_16x16x32_f16(false, a0, false, b0, (short)0, c, false, false);
      c = __builtin_amdgcn_wmma_f32_16x16x32_f16(false, a1, false, b1, (short)0, c, false, false);
#pragma unroll
      for (int r = 0; r < 8; ++r) {
        float x = silu_f(c[r] * 0.125f);           // 1/sqrt(64)
        sH[wave][r + 8 * hlf][nt * 16 + m] = (_Float16)x;
      }
    }
  }

  // ------------------------------------------------------------------
  // Layer 3: H3 (16x64) @ w3 (64x640); wave w covers n-tiles w*5 .. w*5+4.
  // Fold in 1/sqrt(64) and the final 1/sqrt(AVG_NUM_NEIGHBORS).
  // ------------------------------------------------------------------
  {
    v16h a0, a1;
#pragma unroll
    for (int e = 0; e < 16; ++e) {
      const int k = e + 8 * hlf + ((e >= 8) ? 8 : 0);
      a0[e] = sH[wave][m][k];
      a1[e] = sH[wave][m][32 + k];
    }
    const float sc = 0.125f * 0.31622776601683794f;  // (1/8) * 1/sqrt(10)
#pragma unroll
    for (int j = 0; j < 5; ++j) {
      const int nt  = wave * 5 + j;
      const int col = nt * 16 + m;
      v16h b0, b1;
#pragma unroll
      for (int e = 0; e < 16; ++e) {
        const int k = e + 8 * hlf + ((e >= 8) ? 8 : 0);
        b0[e] = (_Float16)w3[k * 640 + col];
        b1[e] = (_Float16)w3[(k + 32) * 640 + col];
      }
      v8f c = {};
      c = __builtin_amdgcn_wmma_f32_16x16x32_f16(false, a0, false, b0, (short)0, c, false, false);
      c = __builtin_amdgcn_wmma_f32_16x16x32_f16(false, a1, false, b1, (short)0, c, false, false);
#pragma unroll
      for (int r = 0; r < 8; ++r)
        sMix[r + 8 * hlf][col] = c[r] * sc;
    }
  }

  __syncthreads();

  // ------------------------------------------------------------------
  // Phase 2: messages + scatter. 16 lanes per edge, 8 channels per lane.
  // ------------------------------------------------------------------
  const int grp = tid >> 4;          // edge within tile (0..15)
  const int t   = tid & 15;
  const int e   = ebase + grp;
  if (e < E) {
    const int snd = senders[e];
    const int rcv = receivers[e];
    const float vx = vectors[3 * e + 0];
    const float vy = vectors[3 * e + 1];
    const float vz = vectors[3 * e + 2];
    const float inr = rsqrtf(vx * vx + vy * vy + vz * vz);
    const float r0 = vx * inr, r1 = vy * inr, r2 = vz * inr;

    const float SH_L1  = 1.7320508075688772f;   // sqrt(3)
    const float SH_L2  = 2.7386127875258306f;   // sqrt(7.5)
    const float CG_011 = 0.5773502691896258f;   // 1/sqrt(3)
    const float CG_110 = 0.5773502691896258f;
    const float CG_121 = 0.6324555320336759f;   // sqrt(0.4)
    const float third  = 1.0f / 3.0f;

    const float Y10 = SH_L1 * r0, Y11 = SH_L1 * r1, Y12 = SH_L1 * r2;
    const float Y200 = SH_L2 * (r0 * r0 - third);
    const float Y201 = SH_L2 * (r0 * r1);
    const float Y202 = SH_L2 * (r0 * r2);
    const float Y211 = SH_L2 * (r1 * r1 - third);
    const float Y212 = SH_L2 * (r1 * r2);
    const float Y222 = SH_L2 * (r2 * r2 - third);

    const float* __restrict__ nf   = node_feats + (size_t)snd * (4 * CCH);
    float*       __restrict__ orow = out        + (size_t)rcv * (11 * CCH);  // 1408
    const float* mixr = sMix[grp];

#pragma unroll
    for (int cc = 0; cc < 8; ++cc) {
      const int ch = t + 16 * cc;
      const float ssc = nf[ch];
      const float v0  = nf[CCH + 3 * ch + 0];
      const float v1  = nf[CCH + 3 * ch + 1];
      const float v2  = nf[CCH + 3 * ch + 2];

      const float tp0 = CG_110 * (v0 * Y10 + v1 * Y11 + v2 * Y12);
      const float t10 = CG_011 * ssc * Y10;
      const float t11 = CG_011 * ssc * Y11;
      const float t12 = CG_011 * ssc * Y12;
      const float t20 = CG_121 * (Y200 * v0 + Y201 * v1 + Y202 * v2);
      const float t21 = CG_121 * (Y201 * v0 + Y211 * v1 + Y212 * v2);
      const float t22 = CG_121 * (Y202 * v0 + Y212 * v1 + Y222 * v2);

      const float ms  = mixr[ch];
      const float mt0 = mixr[CCH + ch];
      const float mv  = mixr[2 * CCH + ch];
      const float mt1 = mixr[3 * CCH + ch];
      const float mt2 = mixr[4 * CCH + ch];

      atomicAdd(&orow[ch],        ssc * ms);
      atomicAdd(&orow[CCH + ch],  tp0 * mt0);

      const int bv  = 2 * CCH + 3 * ch;               // vs block
      atomicAdd(&orow[bv + 0], v0 * mv);
      atomicAdd(&orow[bv + 1], v1 * mv);
      atomicAdd(&orow[bv + 2], v2 * mv);

      const int b1i = 2 * CCH + 3 * (CCH + ch);       // tp1 block
      atomicAdd(&orow[b1i + 0], t10 * mt1);
      atomicAdd(&orow[b1i + 1], t11 * mt1);
      atomicAdd(&orow[b1i + 2], t12 * mt1);

      const int b2i = 2 * CCH + 3 * (2 * CCH + ch);   // tp2 block
      atomicAdd(&orow[b2i + 0], t20 * mt2);
      atomicAdd(&orow[b2i + 1], t21 * mt2);
      atomicAdd(&orow[b2i + 2], t22 * mt2);
    }
  }
}

extern "C" void kernel_launch(void* const* d_in, const int* in_sizes, int n_in,
                              void* d_out, int out_size, void* d_ws, size_t ws_size,
                              hipStream_t stream) {
  (void)n_in; (void)d_ws; (void)ws_size;
  const float* vectors    = (const float*)d_in[0];
  const float* node_feats = (const float*)d_in[1];
  const float* radial     = (const float*)d_in[2];
  const float* w0         = (const float*)d_in[3];
  const float* w1         = (const float*)d_in[4];
  const float* w2         = (const float*)d_in[5];
  const float* w3         = (const float*)d_in[6];
  const int*   senders    = (const int*)d_in[7];
  const int*   receivers  = (const int*)d_in[8];
  float*       out        = (float*)d_out;

  const int E = in_sizes[0] / 3;
  const int R = (E > 0) ? (in_sizes[2] / E) : 8;

  // Output is accumulated with atomics: re-zero every call (deterministic).
  if (out_size > 0)
    zero_out_kernel<<<(out_size + 255) / 256, 256, 0, stream>>>(out, out_size);

  if (E > 0)
    mpconv_kernel<<<(E + 15) / 16, 256, 0, stream>>>(
        vectors, node_feats, radial, w0, w1, w2, w3, senders, receivers, out, E, R);
}
